// CausalSelfAttention_6889127543444
// MI455X (gfx1250) — matrix operand order repaired and checked by reading, not compiled, not run on hardware
//
#include <hip/hip_runtime.h>
#include <hip/hip_bf16.h>
#include <stdint.h>

// ---------------------------------------------------------------------------
// Causal self-attention for MI455X (gfx1250), all matmuls via
// V_WMMA_F32_16X16X32_BF16 (fp32 accumulate). wave32 only.
// A-operand staged to LDS with GLOBAL_LOAD_ASYNC_TO_LDS_B128 (ASYNCcnt).
// Softmax row reductions via DPP16 (quad_perm / row_xmask), no LDS traffic.
// ---------------------------------------------------------------------------

typedef __attribute__((ext_vector_type(16))) __bf16 v16bf;
typedef __attribute__((ext_vector_type(8)))  __bf16 v8bf;
typedef __attribute__((ext_vector_type(8)))  float  v8f;

#define N_EMBD 1024
#define N_HEAD 16
#define HEAD_DIM 64
#define BLOCK_T 2048
#define BATCH 2
#define MROWS (BATCH * BLOCK_T)   // 4096

// round-to-nearest-even f32 -> bf16
static __device__ __forceinline__ __bf16 f2bf(float x) {
  uint32_t u = __builtin_bit_cast(uint32_t, x);
  u += 0x7FFFu + ((u >> 16) & 1u);
  unsigned short h = (unsigned short)(u >> 16);
  return __builtin_bit_cast(__bf16, h);
}

// Build a 16-elem bf16 fragment from two contiguous 16-byte chunks.
static __device__ __forceinline__ v16bf ld_frag(const __bf16* p0, const __bf16* p1) {
  v8bf lo = *(const v8bf*)p0;
  v8bf hi = *(const v8bf*)p1;
  return __builtin_shufflevector(lo, hi, 0, 1, 2, 3, 4, 5, 6, 7,
                                         8, 9, 10, 11, 12, 13, 14, 15);
}

// ---- DPP16 butterfly helpers: reduce over the 16-lane row groups ----------
// xor1 = quad_perm(1,0,3,2)=0xB1, xor2 = quad_perm(2,3,0,1)=0x4E,
// xor4 = ROW_XMASK:4 = 0x164,     xor8 = ROW_XMASK:8 = 0x168
template <int CTRL>
static __device__ __forceinline__ float dppmov(float x) {
  return __builtin_bit_cast(float,
      __builtin_amdgcn_update_dpp(0, __builtin_bit_cast(int, x),
                                  CTRL, 0xf, 0xf, true));
}
static __device__ __forceinline__ float rowmax16(float x) {
  x = fmaxf(x, dppmov<0xB1>(x));
  x = fmaxf(x, dppmov<0x4E>(x));
  x = fmaxf(x, dppmov<0x164>(x));
  x = fmaxf(x, dppmov<0x168>(x));
  return x;
}
static __device__ __forceinline__ float rowsum16(float x) {
  x += dppmov<0xB1>(x);
  x += dppmov<0x4E>(x);
  x += dppmov<0x164>(x);
  x += dppmov<0x168>(x);
  return x;
}

// ---------------------------------------------------------------------------
// x fp32 -> bf16 (straight copy-convert), 4 elements / thread
// ---------------------------------------------------------------------------
__global__ void k_convert(const float* __restrict__ in, __bf16* __restrict__ out, int n) {
  int i = (blockIdx.x * blockDim.x + threadIdx.x) * 4;
  if (i + 3 < n) {
    float4 v = *(const float4*)(in + i);
    out[i + 0] = f2bf(v.x);
    out[i + 1] = f2bf(v.y);
    out[i + 2] = f2bf(v.z);
    out[i + 3] = f2bf(v.w);
  }
}

// ---------------------------------------------------------------------------
// W[r][c] fp32 -> WT[c][r] bf16   (1024x1024), LDS 32x32 tiles
// ---------------------------------------------------------------------------
__global__ void k_transpose(const float* __restrict__ in, __bf16* __restrict__ out,
                            int rows, int cols) {
  __shared__ float tile[32][33];
  int x = blockIdx.x * 32 + threadIdx.x;
  int y0 = blockIdx.y * 32;
  for (int i = threadIdx.y; i < 32; i += 8)
    tile[i][threadIdx.x] = in[(size_t)(y0 + i) * cols + x];
  __syncthreads();
  int xo = blockIdx.y * 32 + threadIdx.x;
  for (int i = threadIdx.y; i < 32; i += 8)
    out[(size_t)(blockIdx.x * 32 + i) * rows + xo] = f2bf(tile[threadIdx.x][i]);
}

// ---------------------------------------------------------------------------
// Generic bf16 GEMM:  C[M][N] = A[M][K] * BT[N][K]^T   (BT pre-transposed)
// Block = 8 waves sharing one 16-row A stripe; each wave owns a 16x64 tile.
// A tile (16x32 bf16 = 1 KB / k-step) is staged into LDS by wave 0 with
// async loads (double buffered); B fragments stream from L2.
// MODE 0: store bf16 row-major [M][N]
// MODE 1: store bf16 V transposed per head: VT[b][h][d][t]
// MODE 2: store fp32 row-major [M][N]
// ---------------------------------------------------------------------------
template <int MODE>
__global__ void k_gemm(const __bf16* __restrict__ A, const __bf16* __restrict__ BT,
                       void* __restrict__ Cout, int M, int N, int K) {
  __shared__ __align__(16) __bf16 asb[2][16][32];   // double-buffered A tile

  const int lane = threadIdx.x & 31;
  const int wvid = threadIdx.x >> 5;
  const int gw   = blockIdx.x * (blockDim.x >> 5) + wvid;
  const int nt64 = N >> 6;
  const int mt   = gw / nt64;        // same for all 8 waves in the block
  const int n64  = gw % nt64;

  const int hl  = lane >> 4;         // 16-lane half
  const int l16 = lane & 15;

  const __bf16* Abase = A + (size_t)(mt * 16) * K;

  // wave 0 stages A[mt*16..+15][kk..kk+31] into asb[bi] (2 async b128 / lane)
  auto stage = [&](int kk, int bi) {
    if (threadIdx.x < 32) {
      int ln = threadIdx.x;
      int r0 = ln >> 2, c0 = (ln & 3) * 8;       // 16-byte chunk
      unsigned long long g0 =
          (unsigned long long)(uintptr_t)(Abase + (size_t)r0 * K + kk + c0);
      unsigned int l0 = (unsigned int)(uintptr_t)&asb[bi][r0][c0];
      asm volatile("global_load_async_to_lds_b128 %0, %1, off"
                   :: "v"(l0), "v"(g0) : "memory");
      int r1 = r0 + 8;
      unsigned long long g1 =
          (unsigned long long)(uintptr_t)(Abase + (size_t)r1 * K + kk + c0);
      unsigned int l1 = (unsigned int)(uintptr_t)&asb[bi][r1][c0];
      asm volatile("global_load_async_to_lds_b128 %0, %1, off"
                   :: "v"(l1), "v"(g1) : "memory");
    }
  };

  v8f acc[4] = {};

  stage(0, 0);
  if (threadIdx.x < 32) asm volatile("s_wait_asynccnt 0x0" ::: "memory");
  __syncthreads();

  for (int kk = 0; kk < K; kk += 32) {
    const int cur = (kk >> 5) & 1;
    if (kk + 32 < K) stage(kk + 32, cur ^ 1);    // prefetch next tile

    // A fragment from LDS (two ds_load_b128)
    v16bf af = ld_frag(&asb[cur][l16][hl << 3],
                       &asb[cur][l16][16 + (hl << 3)]);

#pragma unroll
    for (int nt = 0; nt < 4; ++nt) {
      const __bf16* Brow = BT + (size_t)(n64 * 64 + nt * 16 + l16) * K
                              + kk + (hl << 3);
      v16bf bfr = ld_frag(Brow, Brow + 16);
      acc[nt] = __builtin_amdgcn_wmma_f32_16x16x32_bf16(
          false, af, false, bfr, (short)0, acc[nt], false, false);
    }

    if (threadIdx.x < 32) asm volatile("s_wait_asynccnt 0x0" ::: "memory");
    __syncthreads();
  }

#pragma unroll
  for (int nt = 0; nt < 4; ++nt) {
#pragma unroll
    for (int j = 0; j < 8; ++j) {
      int m = mt * 16 + j + (hl << 3);
      int n = n64 * 64 + nt * 16 + l16;
      float v = acc[nt][j];
      if (MODE == 0) {
        ((__bf16*)Cout)[(size_t)m * N + n] = f2bf(v);
      } else if (MODE == 1) {
        int b = m >> 11, t = m & 2047;
        int h = n >> 6,  d = n & 63;
        ((__bf16*)Cout)[(((size_t)((b * N_HEAD + h) * HEAD_DIM + d)) << 11) + t] = f2bf(v);
      } else {
        ((float*)Cout)[(size_t)m * N + n] = v;
      }
    }
  }
}

// ---------------------------------------------------------------------------
// Flash attention: one wave = one (b,h, 16-row q-tile). kv chunked by 32.
// Qb,Kb: bf16 [4096][1024] row-major (cols = h*64+d)
// VT:    bf16 [B][H][D][T]
// att:   bf16 [4096][1024] row-major
// ---------------------------------------------------------------------------
__global__ void k_attn(const __bf16* __restrict__ Qb, const __bf16* __restrict__ Kb,
                       const __bf16* __restrict__ VT, __bf16* __restrict__ att) {
  __shared__ __align__(16) __bf16 plds[8][16 * 32];   // per-wave 16x32 P tile

  const int lane = threadIdx.x & 31;
  const int wvid = threadIdx.x >> 5;
  const int gw   = blockIdx.x * 8 + wvid;
  const int bh   = gw >> 7;             // 0..31
  const int qt   = gw & 127;            // q tile 0..127
  const int b    = bh >> 4;
  const int h    = bh & 15;
  const int hl   = lane >> 4;
  const int l16  = lane & 15;

  // Q fragments for the two 32-wide halves of D=64
  const __bf16* Qrow = Qb + (size_t)(b * BLOCK_T + qt * 16 + l16) * N_EMBD + h * HEAD_DIM;
  v16bf aq[2];
#pragma unroll
  for (int s = 0; s < 2; ++s)
    aq[s] = ld_frag(Qrow + s * 32 + (hl << 3), Qrow + s * 32 + 16 + (hl << 3));

  v8f o[4] = {};
  float mrow[8], lrow[8];
#pragma unroll
  for (int j = 0; j < 8; ++j) { mrow[j] = -__builtin_inff(); lrow[j] = 0.0f; }

  const int nchunk = (qt >> 1) + 1;     // kv chunks of 32, causal bound
  for (int c = 0; c < nchunk; ++c) {
    const int kv0 = c * 32;

    // ---- scores: S[16][32] = Q(16x64) * K_chunk^T, two 16-col tiles ----
    v8f sacc[2] = {};
#pragma unroll
    for (int t = 0; t < 2; ++t) {
      const __bf16* Krow = Kb + (size_t)(b * BLOCK_T + kv0 + t * 16 + l16) * N_EMBD
                              + h * HEAD_DIM;
#pragma unroll
      for (int s = 0; s < 2; ++s) {
        v16bf bfr = ld_frag(Krow + s * 32 + (hl << 3),
                            Krow + s * 32 + 16 + (hl << 3));
        sacc[t] = __builtin_amdgcn_wmma_f32_16x16x32_bf16(
            false, aq[s], false, bfr, (short)0, sacc[t], false, false);
      }
    }

    // ---- mask + online softmax; P (bf16) -> LDS in [M][col] layout ----
#pragma unroll
    for (int j = 0; j < 8; ++j) {
      const int qg = qt * 16 + j + (hl << 3);
      float s0 = sacc[0][j] * 0.125f;                 // 1/sqrt(64)
      float s1 = sacc[1][j] * 0.125f;
      if (kv0 + l16 > qg)      s0 = -__builtin_inff();
      if (kv0 + 16 + l16 > qg) s1 = -__builtin_inff();

      float mj   = rowmax16(fmaxf(s0, s1));           // DPP16 butterfly
      float mnew = fmaxf(mrow[j], mj);
      float corr = __expf(mrow[j] - mnew);            // exp(-inf - finite) = 0
      mrow[j] = mnew;

      float p0 = __expf(s0 - mnew);
      float p1 = __expf(s1 - mnew);
      float rs = rowsum16(p0 + p1);                   // DPP16 butterfly
      lrow[j] = lrow[j] * corr + rs;

      const int M = j + (hl << 3);
      plds[wvid][(M << 5) + l16]      = f2bf(p0);
      plds[wvid][(M << 5) + 16 + l16] = f2bf(p1);

#pragma unroll
      for (int dt = 0; dt < 4; ++dt) o[dt][j] *= corr;
    }

    // DS ops from one wave are in-order; explicit wait for clarity/safety.
    asm volatile("s_wait_dscnt 0x0" ::: "memory");

    // ---- reload P from LDS as a 16x32 A fragment (two ds_load_b128) ----
    v16bf pf = ld_frag(&plds[wvid][(l16 << 5) + (hl << 3)],
                       &plds[wvid][(l16 << 5) + (hl << 3) + 16]);

    // ---- O(16x64) += P(16x32) * V_chunk(32x64) ----
#pragma unroll
    for (int dt = 0; dt < 4; ++dt) {
      const __bf16* Vrow = VT + (((size_t)(bh * HEAD_DIM + dt * 16 + l16)) << 11)
                              + kv0 + (hl << 3);
      v16bf bfr = ld_frag(Vrow, Vrow + 16);
      o[dt] = __builtin_amdgcn_wmma_f32_16x16x32_bf16(
          false, pf, false, bfr, (short)0, o[dt], false, false);
    }
  }

  // ---- normalize + store att[b*T+q][h*64+d] ----
#pragma unroll
  for (int j = 0; j < 8; ++j) {
    float inv = 1.0f / lrow[j];
    int m = qt * 16 + j + (hl << 3);
#pragma unroll
    for (int dt = 0; dt < 4; ++dt) {
      int d = dt * 16 + l16;
      att[(size_t)(b * BLOCK_T + m) * N_EMBD + h * HEAD_DIM + d] = f2bf(o[dt][j] * inv);
    }
  }
}

// ---------------------------------------------------------------------------
// Launch: convert -> QKV GEMMs -> attention -> proj1 -> proj2
// Workspace layout (bf16 buffers), 42 MB total with aliasing:
//   [0,8)MB   xb   (later reused as att)
//   [8,18)MB  wqT,wkT,wvT,wp1T,wp2T  (2 MB each)
//   [18,26)MB Qb   (later reused as y)
//   [26,34)MB Kb
//   [34,42)MB VT
// ---------------------------------------------------------------------------
extern "C" void kernel_launch(void* const* d_in, const int* in_sizes, int n_in,
                              void* d_out, int out_size, void* d_ws, size_t ws_size,
                              hipStream_t stream) {
  (void)in_sizes; (void)n_in; (void)out_size; (void)ws_size;
  const float* x   = (const float*)d_in[0];
  const float* wq  = (const float*)d_in[1];
  const float* wk  = (const float*)d_in[2];
  const float* wv  = (const float*)d_in[3];
  const float* wp1 = (const float*)d_in[4];
  const float* wp2 = (const float*)d_in[5];
  float* out = (float*)d_out;

  char* ws = (char*)d_ws;
  const size_t MB = 1u << 20;
  __bf16* xb   = (__bf16*)(ws + 0 * MB);
  __bf16* wqT  = (__bf16*)(ws + 8 * MB);
  __bf16* wkT  = (__bf16*)(ws + 10 * MB);
  __bf16* wvT  = (__bf16*)(ws + 12 * MB);
  __bf16* wp1T = (__bf16*)(ws + 14 * MB);
  __bf16* wp2T = (__bf16*)(ws + 16 * MB);
  __bf16* Qb   = (__bf16*)(ws + 18 * MB);
  __bf16* Kb   = (__bf16*)(ws + 26 * MB);
  __bf16* VT   = (__bf16*)(ws + 34 * MB);
  __bf16* att  = xb;   // x dead after QKV GEMMs
  __bf16* y    = Qb;   // Q dead after attention

  // x -> bf16
  k_convert<<<(MROWS * N_EMBD) / (256 * 4), 256, 0, stream>>>(x, xb, MROWS * N_EMBD);

  // weights -> bf16 transposed
  dim3 tb(32, 8), tg(N_EMBD / 32, N_EMBD / 32);
  k_transpose<<<tg, tb, 0, stream>>>(wq,  wqT,  N_EMBD, N_EMBD);
  k_transpose<<<tg, tb, 0, stream>>>(wk,  wkT,  N_EMBD, N_EMBD);
  k_transpose<<<tg, tb, 0, stream>>>(wv,  wvT,  N_EMBD, N_EMBD);
  k_transpose<<<tg, tb, 0, stream>>>(wp1, wp1T, N_EMBD, N_EMBD);
  k_transpose<<<tg, tb, 0, stream>>>(wp2, wp2T, N_EMBD, N_EMBD);

  // QKV projections (4096 wave-tiles of 16x64, 8 waves/block)
  const int gemm_blocks = (MROWS / 16) * (N_EMBD / 64) / 8;   // 512
  k_gemm<0><<<gemm_blocks, 256, 0, stream>>>(xb, wqT, Qb, MROWS, N_EMBD, N_EMBD);
  k_gemm<0><<<gemm_blocks, 256, 0, stream>>>(xb, wkT, Kb, MROWS, N_EMBD, N_EMBD);
  k_gemm<1><<<gemm_blocks, 256, 0, stream>>>(xb, wvT, VT, MROWS, N_EMBD, N_EMBD);

  // attention: 32 (b,h) * 128 q-tiles = 4096 waves
  k_attn<<<512, 256, 0, stream>>>(Qb, Kb, VT, att);

  // output projections
  k_gemm<0><<<gemm_blocks, 256, 0, stream>>>(att, wp1T, y, MROWS, N_EMBD, N_EMBD);
  k_gemm<2><<<gemm_blocks, 256, 0, stream>>>(y, wp2T, out, MROWS, N_EMBD, N_EMBD);
}
